// MultiHeadAttentionLayer_7713761263875
// MI455X (gfx1250) — matrix operand order repaired and checked by reading, not compiled, not run on hardware
//
#include <hip/hip_runtime.h>

typedef unsigned short u16;
typedef __attribute__((ext_vector_type(16))) __bf16 v16bf;
typedef __attribute__((ext_vector_type(8)))  float  v8f;
typedef __attribute__((ext_vector_type(4)))  unsigned int u32x4;
typedef int v4i __attribute__((vector_size(16)));   // matches builtin's vector type

#define Bc  2
#define Sc  2048
#define Dc  1024
#define Hc  16
#define DKc 64
#define Mc  (Bc * Sc)          // 4096 token rows
#define NEG_HUGE (-3.402823466e+38f)

// gfx1250 async global->LDS copy path (ASYNCcnt), if the toolchain exposes it.
#if defined(__has_builtin)
#if __has_builtin(__builtin_amdgcn_global_load_async_to_lds_b128) && \
    __has_builtin(__builtin_amdgcn_s_wait_asynccnt)
#define USE_ASYNC_LDS 1
#endif
#endif
#ifndef USE_ASYNC_LDS
#define USE_ASYNC_LDS 0
#endif

typedef __attribute__((address_space(1))) v4i g_v4i;   // global  (__device__)
typedef __attribute__((address_space(3))) v4i l_v4i;   // LDS     (__shared__)

union V16BF {
    v16bf v;
    u32x4 q[2];
    u16   s[16];
};

__device__ __forceinline__ u16 f32_to_bf16(float f) {
    union { float f; unsigned u; } x{f};
    unsigned r = x.u + 0x7FFFu + ((x.u >> 16) & 1u);   // round-to-nearest-even
    return (u16)(r >> 16);
}

// ---------------- elementwise fp32 -> bf16 convert ----------------
__global__ void k_cvt(const float* __restrict__ src, u16* __restrict__ dst, int n) {
    int i = blockIdx.x * blockDim.x + threadIdx.x;
    int st = gridDim.x * blockDim.x;
    for (; i < n; i += st) dst[i] = f32_to_bf16(src[i]);
}

// ---------------- transpose + convert: Wt[n][k] = W[k][n] ----------------
__global__ void k_tr(const float* __restrict__ W, u16* __restrict__ Wt) {
    int i = blockIdx.x * blockDim.x + threadIdx.x;
    if (i < Dc * Dc) {
        int n = i / Dc, k = i % Dc;
        Wt[i] = f32_to_bf16(W[k * Dc + n]);
    }
}

__device__ __forceinline__ V16BF load_frag(const u16* p) {
    V16BF f;
    f.q[0] = *(const u32x4*)p;
    f.q[1] = *(const u32x4*)(p + 8);
    return f;
}

// copy one 16-byte chunk global -> LDS (async if available)
__device__ __forceinline__ void stage16(u16* ldst, const u16* gsrc) {
#if USE_ASYNC_LDS
    __builtin_amdgcn_global_load_async_to_lds_b128(
        (g_v4i*)gsrc, (l_v4i*)ldst, 0, 0);
#else
    *(u32x4*)ldst = *(const u32x4*)gsrc;
#endif
}

// stage a 128(n) x 32(k) bf16 B-slab into LDS: 512 chunks, 256 threads x 2
__device__ __forceinline__ void stage_b(u16* dst, const u16* __restrict__ Wt,
                                        int n0, int K_, int kk, int tid) {
#pragma unroll
    for (int j = 0; j < 2; ++j) {
        const int idx = tid + 256 * j;
        const int r = idx >> 2, c = idx & 3;
        stage16(dst + r * 32 + c * 8, Wt + (size_t)(n0 + r) * K_ + kk + c * 8);
    }
}

// ---------------- bf16 WMMA GEMM: Y = A[M,K] * Wt^T + bias ----------------
// Block: 8 waves, 128 rows x 128 cols. B staged in LDS (async double-buffer),
// shared by all waves; A register-double-buffered from global.
// mode 0: write bf16 [B,H,S,DK]   (Q / K layout)
// mode 1: write bf16 [B,H,DK,S]   (V transposed layout)
// mode 2: write f32  [M,N]        (final output)
__global__ __launch_bounds__(256)
void k_gemm(const u16* __restrict__ Abf, const u16* __restrict__ Wt,
            const float* __restrict__ bias, void* __restrict__ outp,
            int K_, int N_, int mode) {
    __shared__ __align__(16) u16 bsh[2][128 * 32];   // 2 x 8KB B slabs

    const int tid   = threadIdx.x;
    const int lane  = tid & 31;
    const int wave  = tid >> 5;
    const int m0    = blockIdx.y * 128 + wave * 16;
    const int n0    = blockIdx.x * 128;
    const int row   = lane & 15;
    const int kbA   = (lane < 16) ? 0 : 16;   // A fragment K-base
    const int kbB   = (lane < 16) ? 0 : 16;   // B fragment K-base
    const int mhalf = (lane >> 4) * 8;

    v8f acc[8];
#pragma unroll
    for (int t = 0; t < 8; ++t) acc[t] = (v8f){0,0,0,0,0,0,0,0};

    const u16* arow = Abf + (size_t)(m0 + row) * K_ + kbA;

    // prologue: slab 0 staging + first A fragment in flight
    stage_b(&bsh[0][0], Wt, n0, K_, 0, tid);
    V16BF a_cur = load_frag(arow);
    int buf = 0;

    for (int k0 = 0; k0 < K_; k0 += 32) {
        const int more = (k0 + 32 < K_);
        if (more) stage_b(&bsh[buf ^ 1][0], Wt, n0, K_, k0 + 32, tid);
        const int k0n = more ? k0 + 32 : 0;   // wrapped (in-bounds, unused)
        V16BF a_nxt = load_frag(arow + k0n);
        if (k0 + 128 < K_) __builtin_prefetch(arow + k0 + 128, 0, 1);
#if USE_ASYNC_LDS
        // current slab's copies done; next slab's (2 per thread) may stay in flight
        if (more) __builtin_amdgcn_s_wait_asynccnt(2);
        else      __builtin_amdgcn_s_wait_asynccnt(0);
#endif
        __syncthreads();

        // rolling B fragment out of LDS: issue next before consuming current
        const u16* bbase = &bsh[buf][0];
        V16BF b_cur;
        b_cur.q[0] = *(const u32x4*)(bbase + row * 32 + kbB);
        b_cur.q[1] = *(const u32x4*)(bbase + row * 32 + kbB + 8);
#pragma unroll
        for (int t = 0; t < 8; ++t) {
            V16BF b_nxt;
            if (t < 7) {
                const u16* bp = bbase + (16 * (t + 1) + row) * 32 + kbB;
                b_nxt.q[0] = *(const u32x4*)bp;
                b_nxt.q[1] = *(const u32x4*)(bp + 8);
            }
            acc[t] = __builtin_amdgcn_wmma_f32_16x16x32_bf16(
                false, a_cur.v, false, b_cur.v, (short)0, acc[t], false, false);
            if (t < 7) b_cur = b_nxt;
        }
        __syncthreads();   // all waves done reading this slab before it is re-staged
        a_cur = a_nxt;
        buf ^= 1;
    }

#pragma unroll
    for (int t = 0; t < 8; ++t) {
        const int n = n0 + 16 * t + row;
        const float bv = bias[n];
#pragma unroll
        for (int r = 0; r < 8; ++r) {
            const int m = m0 + r + mhalf;
            const float val = acc[t][r] + bv;
            if (mode == 2) {
                ((float*)outp)[(size_t)m * N_ + n] = val;
            } else {
                const int b = m / Sc, s = m % Sc;
                const int h = n / DKc, dk = n % DKc;
                size_t idx = (mode == 0)
                    ? (((size_t)(b * Hc + h)) * Sc + s) * DKc + dk
                    : (((size_t)(b * Hc + h)) * DKc + dk) * Sc + s;
                ((u16*)outp)[idx] = f32_to_bf16(val);
            }
        }
    }
}

// ---------------- flash attention (per wave: 16 queries, online softmax) ---
__global__ __launch_bounds__(128)
void k_flash(const u16* __restrict__ qbf, const u16* __restrict__ kbf,
             const u16* __restrict__ vtbf, const int* __restrict__ mask,
             u16* __restrict__ obf) {
    __shared__ __align__(16) u16 pshm[4][16 * 32];

    const int lane  = threadIdx.x & 31;
    const int wave  = threadIdx.x >> 5;
    const int bh    = blockIdx.y;
    const int b     = bh / Hc;
    const int h     = bh % Hc;
    const int q0    = blockIdx.x * 64 + wave * 16;
    const int row   = lane & 15;
    const int kbA   = (lane < 16) ? 0 : 16;
    const int kbB   = (lane < 16) ? 0 : 16;
    const int mhalf = (lane >> 4) * 8;

    // Q tile (16 x 64) as two 16x32 A fragments, loaded once
    const u16* qrow = qbf + ((size_t)bh * Sc + q0 + row) * DKc;
    V16BF aq0, aq1;
    aq0.q[0] = *(const u32x4*)(qrow + kbA);
    aq0.q[1] = *(const u32x4*)(qrow + kbA + 8);
    aq1.q[0] = *(const u32x4*)(qrow + 32 + kbA);
    aq1.q[1] = *(const u32x4*)(qrow + 32 + kbA + 8);

    v8f o[4];
#pragma unroll
    for (int t = 0; t < 4; ++t) o[t] = (v8f){0,0,0,0,0,0,0,0};
    float mrow[8], lrow[8];
#pragma unroll
    for (int r = 0; r < 8; ++r) { mrow[r] = NEG_HUGE; lrow[r] = 0.f; }

    for (int key0 = 0; key0 < Sc; key0 += 32) {
        // ---- issue all K fragments, then all V fragments (loads overlap) ----
        V16BF kf[4];
#pragma unroll
        for (int n = 0; n < 2; ++n) {
            const u16* krow = kbf + ((size_t)bh * Sc + key0 + 16 * n + row) * DKc;
            kf[2 * n].q[0]     = *(const u32x4*)(krow + kbB);
            kf[2 * n].q[1]     = *(const u32x4*)(krow + kbB + 8);
            kf[2 * n + 1].q[0] = *(const u32x4*)(krow + 32 + kbB);
            kf[2 * n + 1].q[1] = *(const u32x4*)(krow + 32 + kbB + 8);
        }
        V16BF vf[4];
#pragma unroll
        for (int t = 0; t < 4; ++t) {
            const u16* vrow = vtbf + ((size_t)bh * DKc + 16 * t + row) * Sc + key0 + kbB;
            vf[t].q[0] = *(const u32x4*)vrow;
            vf[t].q[1] = *(const u32x4*)(vrow + 8);
        }
        const int mz0 = mask[b * Sc + key0 + row];
        const int mz1 = mask[b * Sc + key0 + 16 + row];

        // ---- scores: S = Q * K^T (waits only on K loads; V still in flight) ----
        v8f z = (v8f){0,0,0,0,0,0,0,0};
        v8f sc0 = __builtin_amdgcn_wmma_f32_16x16x32_bf16(
            false, aq0.v, false, kf[0].v, (short)0, z, false, false);
        sc0 = __builtin_amdgcn_wmma_f32_16x16x32_bf16(
            false, aq1.v, false, kf[1].v, (short)0, sc0, false, false);
        v8f sc1 = __builtin_amdgcn_wmma_f32_16x16x32_bf16(
            false, aq0.v, false, kf[2].v, (short)0, z, false, false);
        sc1 = __builtin_amdgcn_wmma_f32_16x16x32_bf16(
            false, aq1.v, false, kf[3].v, (short)0, sc1, false, false);

        // ---- online softmax over the 32 new columns ----
#pragma unroll
        for (int r = 0; r < 8; ++r) {
            float s0 = mz0 ? NEG_HUGE : sc0[r] * 0.125f;
            float s1 = mz1 ? NEG_HUGE : sc1[r] * 0.125f;
            float rm = fmaxf(s0, s1);
            rm = fmaxf(rm, __shfl_xor(rm, 1, 16));
            rm = fmaxf(rm, __shfl_xor(rm, 2, 16));
            rm = fmaxf(rm, __shfl_xor(rm, 4, 16));
            rm = fmaxf(rm, __shfl_xor(rm, 8, 16));
            float mn   = fmaxf(mrow[r], rm);
            float corr = __expf(mrow[r] - mn);
            float p0   = __expf(s0 - mn);
            float p1   = __expf(s1 - mn);
            float rs   = p0 + p1;
            rs += __shfl_xor(rs, 1, 16);
            rs += __shfl_xor(rs, 2, 16);
            rs += __shfl_xor(rs, 4, 16);
            rs += __shfl_xor(rs, 8, 16);
            lrow[r] = lrow[r] * corr + rs;
            mrow[r] = mn;
#pragma unroll
            for (int t = 0; t < 4; ++t) o[t][r] *= corr;
            // stash P (C-layout) into wave-private LDS as bf16
            u16* pr = &pshm[wave][(r + mhalf) * 32 + row];
            pr[0]  = f32_to_bf16(p0);
            pr[16] = f32_to_bf16(p1);
        }

        // Wave-private LDS region; DS ops from one wave complete in order, so a
        // workgroup barrier is unnecessary -- only compiler code motion must be
        // fenced across the cross-lane store->load.
        __builtin_amdgcn_fence(__ATOMIC_RELEASE, "wavefront");
        __builtin_amdgcn_wave_barrier();
        V16BF pa;
        const u16* pl = &pshm[wave][row * 32];
        pa.q[0] = *(const u32x4*)(pl + kbA);
        pa.q[1] = *(const u32x4*)(pl + kbA + 8);
        __builtin_amdgcn_wave_barrier();
        __builtin_amdgcn_fence(__ATOMIC_ACQUIRE, "wavefront");

        // ---- O += P * V  (V fragments already resident) ----
#pragma unroll
        for (int t = 0; t < 4; ++t) {
            o[t] = __builtin_amdgcn_wmma_f32_16x16x32_bf16(
                false, pa.v, false, vf[t].v, (short)0, o[t], false, false);
        }
    }

    // ---- normalize and write O to row-major [B*S, D] (bf16, head-concat) ----
#pragma unroll
    for (int t = 0; t < 4; ++t) {
#pragma unroll
        for (int r = 0; r < 8; ++r) {
            float val = o[t][r] / lrow[r];
            int sq  = q0 + r + mhalf;
            int col = h * DKc + 16 * t + row;
            obf[((size_t)b * Sc + sq) * Dc + col] = f32_to_bf16(val);
        }
    }
}

extern "C" void kernel_launch(void* const* d_in, const int* in_sizes, int n_in,
                              void* d_out, int out_size, void* d_ws, size_t ws_size,
                              hipStream_t stream) {
    (void)in_sizes; (void)n_in; (void)out_size; (void)ws_size;
    const float* q    = (const float*)d_in[0];
    const float* k    = (const float*)d_in[1];
    const float* v    = (const float*)d_in[2];
    const int*   mask = (const int*)  d_in[3];
    const float* Wq   = (const float*)d_in[4];
    const float* bq   = (const float*)d_in[5];
    const float* Wk   = (const float*)d_in[6];
    const float* bk   = (const float*)d_in[7];
    const float* Wv   = (const float*)d_in[8];
    const float* bv   = (const float*)d_in[9];
    const float* Wo   = (const float*)d_in[10];
    const float* bo   = (const float*)d_in[11];

    const size_t MD = (size_t)Mc * Dc;      // 4096*1024
    const size_t DD = (size_t)Dc * Dc;
    u16* ws  = (u16*)d_ws;
    u16* xq  = ws;                 // bf16 query activations (reused for O later)
    u16* xk  = xq  + MD;
    u16* xv  = xk  + MD;
    u16* wqt = xv  + MD;
    u16* wkt = wqt + DD;
    u16* wvt = wkt + DD;
    u16* wot = wvt + DD;
    u16* qb  = wot + DD;           // Q  [B,H,S,DK] bf16
    u16* kb  = qb  + MD;           // K  [B,H,S,DK] bf16
    u16* vt  = kb  + MD;           // V^T [B,H,DK,S] bf16
    u16* ob  = xq;                 // O  [B*S, D] bf16 (reuse xq slot)

    k_cvt<<<2048, 256, 0, stream>>>(q, xq, (int)MD);
    k_cvt<<<2048, 256, 0, stream>>>(k, xk, (int)MD);
    k_cvt<<<2048, 256, 0, stream>>>(v, xv, (int)MD);
    k_tr <<<(Dc * Dc) / 256, 256, 0, stream>>>(Wq, wqt);
    k_tr <<<(Dc * Dc) / 256, 256, 0, stream>>>(Wk, wkt);
    k_tr <<<(Dc * Dc) / 256, 256, 0, stream>>>(Wv, wvt);
    k_tr <<<(Dc * Dc) / 256, 256, 0, stream>>>(Wo, wot);

    dim3 gg(Dc / 128, Mc / 128);
    k_gemm<<<gg, 256, 0, stream>>>(xq, wqt, bq, qb, Dc, Dc, 0);
    k_gemm<<<gg, 256, 0, stream>>>(xk, wkt, bk, kb, Dc, Dc, 0);
    k_gemm<<<gg, 256, 0, stream>>>(xv, wvt, bv, vt, Dc, Dc, 1);

    dim3 gf(Sc / 64, Bc * Hc);
    k_flash<<<gf, 128, 0, stream>>>(qb, kb, vt, mask, ob);

    k_gemm<<<gg, 256, 0, stream>>>(ob, wot, bo, d_out, Dc, Dc, 2);
}
